// spatialAttentionScaledGCN_64957085384761
// MI455X (gfx1250) — matrix operand order repaired and checked by reading, not compiled, not run on hardware
//
#include <hip/hip_runtime.h>
#include <math.h>

#define B_  8
#define N_  512
#define T_  12
#define F_  32
#define FO_ 64
#define SCALE 0.17677669529663687f   // 1/sqrt(32)

typedef _Float16 half_t;
typedef half_t v16h __attribute__((ext_vector_type(16)));
typedef half_t v4h  __attribute__((ext_vector_type(4)));
typedef float  v8f  __attribute__((ext_vector_type(8)));

__device__ __forceinline__ v8f wmma_f16(v16h a, v16h b, v8f c) {
  // D = A(16x32) * B(32x16) + C, f16 in / f32 out
  return __builtin_amdgcn_wmma_f32_16x16x32_f16(false, a, false, b, (short)0, c,
                                                false, false);
}

// ---- Async staging: global f16 slice -> LDS, no VGPR round trip ---------
// Copies x[b, :, t, :] (512x32 f16 = 32KB) into LDS row-major [m][f] using
// GLOBAL_LOAD_ASYNC_TO_LDS_B128 (ASYNCcnt-tracked DMA per lane).
__device__ __forceinline__ void stage_slice_async(const half_t* __restrict__ src,
                                                  int b, int t, half_t* lds) {
  const half_t* base = src + (size_t)b * N_ * T_ * F_ + (size_t)t * F_;
  for (int e = threadIdx.x; e < (N_ * F_) / 8; e += 128) {   // 2048 16B chunks
    const int m = e >> 2;                  // 4 x b128 per 32-half row
    const int q = (e & 3) * 8;
    const half_t* g = base + (size_t)m * (T_ * F_) + q;
    const uint32_t l = (uint32_t)(uintptr_t)(lds + m * F_ + q);  // LDS byte offset
    asm volatile("global_load_async_to_lds_b128 %0, %1, off"
                 :: "v"(l), "v"(g) : "memory");
  }
}

__device__ __forceinline__ void wait_async_and_barrier() {
  asm volatile("s_wait_asynccnt 0" ::: "memory");
  __syncthreads();
}

// ---- Fragment loaders (wave32 layouts per CDNA5 ISA 7.12.2) -------------
// A (16x32 f16, MxK) from row-major LDS lds[row*stride + k].
__device__ __forceinline__ v16h load_a_rm(const half_t* lds, int row_base, int stride) {
  const int lane = threadIdx.x & 31;
  const half_t* p = lds + (row_base + (lane & 15)) * stride;
  const int h = (lane >> 4) * 8;
  v16h a;
#pragma unroll
  for (int j = 0; j < 8; ++j) a[j] = p[h + j];
#pragma unroll
  for (int j = 0; j < 8; ++j) a[8 + j] = p[16 + h + j];
  return a;
}

// B (32x16 f16, KxN) where B[k][n] = lds[(col_base+n)*stride + k]  (n-major)
__device__ __forceinline__ v16h load_b_nmajor(const half_t* lds, int col_base, int stride) {
  const int lane = threadIdx.x & 31;
  const half_t* p = lds + (col_base + (lane & 15)) * stride + (lane >> 4) * 8;
  v16h b;
#pragma unroll
  for (int j = 0; j < 16; ++j) b[j] = p[(j & 7) + ((j >> 3) << 4)];
  return b;
}

// B (32x16 f16, KxN) where B[k][n] = lds[(k_base+k)*stride + col_base + n] (k-major)
__device__ __forceinline__ v16h load_b_kmajor(const half_t* lds, int k_base, int col_base, int stride) {
  const int lane = threadIdx.x & 31;
  const int n = col_base + (lane & 15);
  const int ks = k_base + (lane >> 4) * 8;
  v16h b;
#pragma unroll
  for (int j = 0; j < 16; ++j) b[j] = lds[(ks + (j & 7) + ((j >> 3) << 4)) * stride + n];
  return b;
}

// ---- init: x (f32) -> both f16 ping-pong buffers -------------------------
__global__ __launch_bounds__(256) void cvt_kernel(const float4* __restrict__ x,
                                                  v4h* __restrict__ b0,
                                                  v4h* __restrict__ b1) {
  const size_t i = (size_t)blockIdx.x * 256 + threadIdx.x;
  const float4 v = x[i];
  v4h h;
  h[0] = (half_t)v.x; h[1] = (half_t)v.y; h[2] = (half_t)v.z; h[3] = (half_t)v.w;
  b0[i] = h;
  b1[i] = h;
}

// ---- Kernel A: per-row logsumexp of scaled attention scores --------------
__global__ __launch_bounds__(128) void rowstats_kernel(const half_t* __restrict__ x,
                                                       float* __restrict__ lse, int tbase) {
  __shared__ half_t xs[N_ * F_];
  const int t = blockIdx.y + tbase;
  const int b = blockIdx.z;
  stage_slice_async(x, b, t, xs);
  wait_async_and_barrier();

  const int wave = threadIdx.x >> 5;
  const int lane = threadIdx.x & 31;
  const int rb = blockIdx.x * 64 + wave * 16;   // 16-row strip per wave
  const v16h a = load_a_rm(xs, rb, F_);

  float mrun[8], srun[8];
#pragma unroll
  for (int r = 0; r < 8; ++r) { mrun[r] = -1e30f; srun[r] = 0.0f; }

  for (int ct = 0; ct < 32; ++ct) {             // sweep all 512 columns
    const v16h bf = load_b_nmajor(xs, ct * 16, F_);
    v8f c = {};
    c = wmma_f16(a, bf, c);
#pragma unroll
    for (int r = 0; r < 8; ++r) {               // online softmax per row slot
      const float v = c[r] * SCALE;
      const float mn = fmaxf(mrun[r], v);
      srun[r] = srun[r] * __expf(mrun[r] - mn) + __expf(v - mn);
      mrun[r] = mn;
    }
  }
#pragma unroll
  for (int r = 0; r < 8; ++r) {
#pragma unroll
    for (int off = 1; off < 16; off <<= 1) {    // reduce across the 16 col-lanes
      const float mo = __shfl_xor(mrun[r], off, 32);
      const float so = __shfl_xor(srun[r], off, 32);
      const float mn = fmaxf(mrun[r], mo);
      srun[r] = srun[r] * __expf(mrun[r] - mn) + so * __expf(mo - mn);
      mrun[r] = mn;
    }
    if ((lane & 15) == 0) {
      const int row = rb + r + 8 * (lane >> 4);
      lse[((size_t)b * T_ + t) * N_ + row] = mrun[r] + __logf(srun[r]);
    }
  }
}

// ---- Kernel B: one fori_loop body iteration ------------------------------
__global__ __launch_bounds__(128) void update_kernel(const half_t* __restrict__ xsrc,
                                                     half_t* __restrict__ xdst,
                                                     const float* __restrict__ phase,
                                                     const float* __restrict__ mask,
                                                     const float* __restrict__ lse) {
  __shared__ half_t xs[N_ * F_];   // x slice at tau   (scores)
  __shared__ half_t xp[N_ * F_];   // x slice at tau-1 (values)
  const int tau = blockIdx.y + 1;
  const int b = blockIdx.z;
  stage_slice_async(xsrc, b, tau, xs);
  stage_slice_async(xsrc, b, tau - 1, xp);
  wait_async_and_barrier();

  const int wave = threadIdx.x >> 5;
  const int lane = threadIdx.x & 31;
  const int n0 = blockIdx.x * 64 + wave * 16;   // 16-column strip per wave
  const int nl = lane & 15;
  const int h8 = (lane >> 4) * 8;
  const int ncol = n0 + nl;
  const float* lse_bt = lse + ((size_t)b * T_ + tau) * N_;
  const float* ph = phase + ((size_t)b * T_ + tau) * N_ * N_;

  const v16h bn = load_b_nmajor(xs, n0, F_);    // fixed: this wave's 16 key columns
  v8f acc0 = {}, acc1 = {};

  for (int kc = 0; kc < 16; ++kc) {             // K = 512 in chunks of 32 rows m
    const int mb = kc * 32;
    const v16h a0 = load_a_rm(xs, mb, F_);
    const v16h a1 = load_a_rm(xs, mb + 16, F_);
    v8f c0 = {}, c1 = {};
    c0 = wmma_f16(a0, bn, c0);                  // scores rows mb..mb+15
    c1 = wmma_f16(a1, bn, c1);                  // scores rows mb+16..mb+31
    if (kc < 15) __builtin_prefetch(ph + (size_t)(mb + 32) * N_ + ncol, 0, 1);

    // D-layout of scores maps lane-exactly onto A-fragment of the value GEMM
    v16h aw;
#pragma unroll
    for (int r = 0; r < 8; ++r) {
      const int m0 = mb + h8 + r;
      const int m1 = mb + 16 + h8 + r;
      const float w0 = ph[(size_t)m0 * N_ + ncol] * __expf(c0[r] * SCALE - lse_bt[m0]) * SCALE;
      const float w1 = ph[(size_t)m1 * N_ + ncol] * __expf(c1[r] * SCALE - lse_bt[m1]) * SCALE;
      aw[r] = (half_t)w0;
      aw[8 + r] = (half_t)w1;
    }
    const v16h b0 = load_b_kmajor(xp, mb, 0, F_);
    const v16h b1 = load_b_kmajor(xp, mb, 16, F_);
    acc0 = wmma_f16(aw, b0, acc0);              // f = 0..15
    acc1 = wmma_f16(aw, b1, acc1);              // f = 16..31
  }

#pragma unroll
  for (int r = 0; r < 8; ++r) {
    const int n = n0 + r + h8;
    const float mk = mask[n];
    const size_t base = (((size_t)b * N_ + n) * T_ + tau) * F_;
    const float old0 = (float)xsrc[base + nl];
    const float old1 = (float)xsrc[base + 16 + nl];
    xdst[base + nl]      = (half_t)(mk * old0 + (1.0f - mk) * acc0[r]);
    xdst[base + 16 + nl] = (half_t)(mk * old1 + (1.0f - mk) * acc1[r]);
  }
}

// ---- Kernel C: final aggregation + theta projection + relu ---------------
__global__ __launch_bounds__(128) void final_kernel(const half_t* __restrict__ xsrc,
                                                    const float* __restrict__ phase,
                                                    const float* __restrict__ adj,
                                                    const float* __restrict__ theta,
                                                    const float* __restrict__ lse,
                                                    float* __restrict__ out) {
  __shared__ half_t xs[N_ * F_];
  __shared__ half_t th[F_ * FO_];
  __shared__ half_t aggs[4][16 * F_];
  const int t = blockIdx.y;
  const int b = blockIdx.z;
  stage_slice_async(xsrc, b, t, xs);
  for (int e = threadIdx.x; e < F_ * FO_; e += 128) th[e] = (half_t)theta[e];
  wait_async_and_barrier();

  const int wave = threadIdx.x >> 5;
  const int lane = threadIdx.x & 31;
  const int n0 = blockIdx.x * 64 + wave * 16;
  const int nl = lane & 15;
  const int h8 = (lane >> 4) * 8;
  const int ncol = n0 + nl;
  const float* lse_bt = lse + ((size_t)b * T_ + t) * N_;
  const float* ph = phase + ((size_t)b * T_ + t) * N_ * N_;

  const v16h bn = load_b_nmajor(xs, n0, F_);
  v8f acc0 = {}, acc1 = {};
  for (int kc = 0; kc < 16; ++kc) {
    const int mb = kc * 32;
    const v16h a0 = load_a_rm(xs, mb, F_);
    const v16h a1 = load_a_rm(xs, mb + 16, F_);
    v8f c0 = {}, c1 = {};
    c0 = wmma_f16(a0, bn, c0);
    c1 = wmma_f16(a1, bn, c1);
    if (kc < 15) __builtin_prefetch(ph + (size_t)(mb + 32) * N_ + ncol, 0, 1);
    v16h aw;
#pragma unroll
    for (int r = 0; r < 8; ++r) {
      const int m0 = mb + h8 + r;
      const int m1 = mb + 16 + h8 + r;
      const float w0 = ph[(size_t)m0 * N_ + ncol] * adj[(size_t)m0 * N_ + ncol]
                       * __expf(c0[r] * SCALE - lse_bt[m0]) * SCALE;
      const float w1 = ph[(size_t)m1 * N_ + ncol] * adj[(size_t)m1 * N_ + ncol]
                       * __expf(c1[r] * SCALE - lse_bt[m1]) * SCALE;
      aw[r] = (half_t)w0;
      aw[8 + r] = (half_t)w1;
    }
    const v16h b0 = load_b_kmajor(xs, mb, 0, F_);   // values = same time slice
    const v16h b1 = load_b_kmajor(xs, mb, 16, F_);
    acc0 = wmma_f16(aw, b0, acc0);
    acc1 = wmma_f16(aw, b1, acc1);
  }

  // Stage agg (16x32) through LDS to re-fragment as A for the theta GEMM
#pragma unroll
  for (int r = 0; r < 8; ++r) {
    aggs[wave][(r + h8) * F_ + nl]      = (half_t)acc0[r];
    aggs[wave][(r + h8) * F_ + 16 + nl] = (half_t)acc1[r];
  }
  __syncthreads();
  const v16h ag = load_a_rm(aggs[wave], 0, F_);
#pragma unroll
  for (int ot = 0; ot < 4; ++ot) {                  // F_OUT = 64 in 4 tiles
    const v16h bt = load_b_kmajor(th, 0, ot * 16, FO_);
    v8f d = {};
    d = wmma_f16(ag, bt, d);
#pragma unroll
    for (int r = 0; r < 8; ++r) {
      const int n = n0 + r + h8;
      const int o = ot * 16 + nl;
      out[(((size_t)b * N_ + n) * T_ + t) * FO_ + o] = fmaxf(d[r], 0.0f);
    }
  }
}

// ---- Host side -----------------------------------------------------------
extern "C" void kernel_launch(void* const* d_in, const int* in_sizes, int n_in,
                              void* d_out, int out_size, void* d_ws, size_t ws_size,
                              hipStream_t stream) {
  const float* x     = (const float*)d_in[0];
  const float* phase = (const float*)d_in[1];
  const float* adj   = (const float*)d_in[2];
  const float* mask  = (const float*)d_in[3];
  const float* theta = (const float*)d_in[4];
  float* out = (float*)d_out;

  const size_t XN = (size_t)B_ * N_ * T_ * F_;      // 1,572,864 elements
  char* ws = (char*)d_ws;
  half_t* buf0 = (half_t*)ws;
  half_t* buf1 = buf0 + XN;
  float*  lse  = (float*)(ws + 2 * XN * sizeof(half_t));   // B*T*N floats

  // init both f16 ping-pong buffers from x (t=0 slice is never rewritten)
  cvt_kernel<<<dim3((unsigned)(XN / 4 / 256)), 256, 0, stream>>>(
      (const float4*)x, (v4h*)buf0, (v4h*)buf1);

  half_t* bufs[2] = {buf0, buf1};
  for (int k = 1; k <= 11; ++k) {                   // fori_loop(1, T)
    const half_t* src = bufs[(k - 1) & 1];
    half_t* dst = bufs[k & 1];
    rowstats_kernel<<<dim3(8, 11, 8), 128, 0, stream>>>(src, lse, 1);
    update_kernel<<<dim3(8, 11, 8), 128, 0, stream>>>(src, dst, phase, mask, lse);
  }
  // final x lives in buf1 (k=11 wrote bufs[1])
  rowstats_kernel<<<dim3(8, 12, 8), 128, 0, stream>>>(buf1, lse, 0);
  final_kernel<<<dim3(8, 12, 8), 128, 0, stream>>>(buf1, phase, adj, theta, lse, out);
}
